// SpikeNet_87024627352088
// MI455X (gfx1250) — matrix operand order repaired and checked by reading, not compiled, not run on hardware
//
#include <hip/hip_runtime.h>
#include <hip/hip_bf16.h>

// ---------------------------------------------------------------------------
// SpikeNet on MI455X (gfx1250), wave32 + WMMA bf16.
//
// TAU == 1.0 makes the LIF recurrence stateless (v is overwritten by a each
// step), so each timestep is: gather/avg -> GEMM(K=512, bf16 WMMA) -> spike.
// Gather traffic dominates (x table ~100MB lives in 192MB L2); WMMA compute
// (~75 GFLOP bf16) is negligible, so everything is fused to minimize passes.
// ---------------------------------------------------------------------------

typedef __bf16 bf16_t;
typedef __bf16 v16bf __attribute__((ext_vector_type(16)));
typedef __bf16 v8bf  __attribute__((ext_vector_type(8)));
typedef __bf16 v4bf  __attribute__((ext_vector_type(4)));
typedef float  v8f   __attribute__((ext_vector_type(8)));

// Problem constants (from reference setup_inputs)
#define NNODES   8192
#define FDIM     256
#define HDIM     256
#define CDIM     64
#define TSTEPS   5
#define S1       5
#define S2       2
#define M0ROWS   (NNODES * (1 + S1))   // 49152
#define KDIM     512                   // concat [self | neigh]

// Workspace layout (bytes); total ~134.7 MB
#define OFF_A0   ((size_t)0)                       // 49152*512*2  = 50331648
#define OFF_S0   ((size_t)50331648)                // 49152*256*2  = 25165824
#define OFF_A1   ((size_t)75497472)                //  8192*512*2  =  8388608
#define OFF_BT0  ((size_t)83886080)                //   256*512*2  =   262144
#define OFF_BT1  ((size_t)84148224)                //   256*512*2  =   262144
#define OFF_BS0  ((size_t)84410368)                //   256*4
#define OFF_BS1  ((size_t)84411392)                //   256*4
#define OFF_SPK  ((size_t)84412416)                // 8192*5*256*4 = 41943040
#define OFF_YM   ((size_t)126355456)               // 8192*256*4   =  8388608

// ---------------------------------------------------------------------------
// Prep: transpose+convert weights into N-major bf16 [256 x 512] so WMMA B
// fragments are contiguous b128 loads; fold the two biases into one vector.
// ---------------------------------------------------------------------------
__global__ __launch_bounds__(256) void prep_kernel(
    const float* __restrict__ Wl0, const float* __restrict__ Wr0,
    const float* __restrict__ Wl1, const float* __restrict__ Wr1,
    const float* __restrict__ bl0, const float* __restrict__ br0,
    const float* __restrict__ bl1, const float* __restrict__ br1,
    bf16_t* __restrict__ Bt0, bf16_t* __restrict__ Bt1,
    float* __restrict__ bs0, float* __restrict__ bs1)
{
    const int nc = blockIdx.x;    // output column 0..255
    const int k  = threadIdx.x;   // inner 0..255
    Bt0[(size_t)nc * KDIM + k]       = (bf16_t)Wl0[(size_t)k * HDIM + nc];
    Bt0[(size_t)nc * KDIM + 256 + k] = (bf16_t)Wr0[(size_t)k * HDIM + nc];
    Bt1[(size_t)nc * KDIM + k]       = (bf16_t)Wl1[(size_t)k * HDIM + nc];
    Bt1[(size_t)nc * KDIM + 256 + k] = (bf16_t)Wr1[(size_t)k * HDIM + nc];
    if (blockIdx.x == 0) {
        bs0[k] = bl0[k] + br0[k];
        bs1[k] = bl1[k] + br1[k];
    }
}

// ---------------------------------------------------------------------------
// Gather + neighbor averaging for layer 0, fused with f32->bf16 convert.
// Builds A0 row r = [ self0[r] | neigh0[r] ]  (49152 x 512 bf16).
// x is 100MB -> mostly L2-resident; float4 loads, 8B bf16x4 stores.
// ---------------------------------------------------------------------------
__global__ __launch_bounds__(64) void gather0_kernel(
    const float* __restrict__ x, const int* __restrict__ nodes,
    const int* __restrict__ nbr1, const int* __restrict__ nbr2,
    bf16_t* __restrict__ A0, int t)
{
    const int row = blockIdx.x;          // 0..49151
    const int col = threadIdx.x * 4;     // 0..252
    float sx, sy, sz, sw;
    float ax = 0.f, ay = 0.f, az = 0.f, aw = 0.f;

    if (row < NNODES) {
        const float4 sv = *(const float4*)(x + (size_t)nodes[row] * FDIM + col);
        sx = sv.x; sy = sv.y; sz = sv.z; sw = sv.w;
        const int* p = nbr1 + (size_t)t * (NNODES * S1) + row * S1;
        #pragma unroll
        for (int i = 0; i < S1; ++i) {
            const float4 v = *(const float4*)(x + (size_t)p[i] * FDIM + col);
            ax += v.x; ay += v.y; az += v.z; aw += v.w;
        }
        ax *= 0.2f; ay *= 0.2f; az *= 0.2f; aw *= 0.2f;
    } else {
        const int j = row - NNODES;      // 0..40959
        const float4 sv = *(const float4*)(x + (size_t)nbr1[(size_t)t * (NNODES * S1) + j] * FDIM + col);
        sx = sv.x; sy = sv.y; sz = sv.z; sw = sv.w;
        const int* p = nbr2 + (size_t)t * (NNODES * S1 * S2) + j * S2;
        #pragma unroll
        for (int i = 0; i < S2; ++i) {
            const float4 v = *(const float4*)(x + (size_t)p[i] * FDIM + col);
            ax += v.x; ay += v.y; az += v.z; aw += v.w;
        }
        ax *= 0.5f; ay *= 0.5f; az *= 0.5f; aw *= 0.5f;
    }

    v4bf s4, n4;
    s4[0] = (bf16_t)sx; s4[1] = (bf16_t)sy; s4[2] = (bf16_t)sz; s4[3] = (bf16_t)sw;
    n4[0] = (bf16_t)ax; n4[1] = (bf16_t)ay; n4[2] = (bf16_t)az; n4[3] = (bf16_t)aw;
    *(v4bf*)(A0 + (size_t)row * KDIM + col)       = s4;
    *(v4bf*)(A0 + (size_t)row * KDIM + 256 + col) = n4;
}

// ---------------------------------------------------------------------------
// Build A1 = [ s0[:N] | mean_{S1}(s0[N:]) ]  (8192 x 512 bf16).
// ---------------------------------------------------------------------------
__global__ __launch_bounds__(64) void build_a1_kernel(
    const bf16_t* __restrict__ s0, bf16_t* __restrict__ A1)
{
    const int row = blockIdx.x;        // 0..8191
    const int col = threadIdx.x * 4;   // 0..252
    *(v4bf*)(A1 + (size_t)row * KDIM + col) =
        *(const v4bf*)(s0 + (size_t)row * HDIM + col);

    float a0 = 0.f, a1 = 0.f, a2 = 0.f, a3 = 0.f;
    #pragma unroll
    for (int i = 0; i < S1; ++i) {
        const v4bf v = *(const v4bf*)(s0 + (size_t)(NNODES + row * S1 + i) * HDIM + col);
        a0 += (float)v[0]; a1 += (float)v[1]; a2 += (float)v[2]; a3 += (float)v[3];
    }
    v4bf n4;
    n4[0] = (bf16_t)(a0 * 0.2f); n4[1] = (bf16_t)(a1 * 0.2f);
    n4[2] = (bf16_t)(a2 * 0.2f); n4[3] = (bf16_t)(a3 * 0.2f);
    *(v4bf*)(A1 + (size_t)row * KDIM + 256 + col) = n4;
}

// ---------------------------------------------------------------------------
// bf16 WMMA GEMM (M x 512) @ (512 x 256) with fused bias + LIF spike.
// One wave computes a 16x64 output strip: 4x v_wmma_f32_16x16x32_bf16 per
// 32-wide K step, reusing the A fragment. Block = 4 waves -> 64x64 tile.
//
// A fragment (16-bit 16x32): lane L, row = L&15; elems 0..7 = K+((L>>4)*8),
// elems 8..15 = K+16+((L>>4)*8). B from pre-transposed Bt[n][k]: lane L,
// n = nbase+(L&15), same K order as A (K+((L>>4)*8), K+16+((L>>4)*8)).
// C layout: VGPR r, lane L -> row = mbase + 8*(L>>4) + r, col = nbase + (L&15).
//
// MODE 0: write bf16 spikes (M x 256).  MODE 1: write f32 into spk[n][t][c].
// ---------------------------------------------------------------------------
template <int MODE>
__global__ __launch_bounds__(128) void gemm_lif_kernel(
    const bf16_t* __restrict__ A,    // M x 512
    const bf16_t* __restrict__ Bt,   // 256 x 512 (N-major)
    const float*  __restrict__ bsum, // 256
    void* __restrict__ outp, int t)
{
    const int wave  = threadIdx.x >> 5;
    const int lane  = threadIdx.x & 31;
    const int half  = lane >> 4;
    const int l16   = lane & 15;
    const int mbase = blockIdx.x * 64 + wave * 16;
    const int nbase = blockIdx.y * 64;

    const size_t arow = (size_t)(mbase + l16) * KDIM;

    v8f acc[4] = {{}, {}, {}, {}};

    for (int k0 = 0; k0 < KDIM; k0 += 32) {
        const v8bf alo = *(const v8bf*)(A + arow + k0 + half * 8);
        const v8bf ahi = *(const v8bf*)(A + arow + k0 + 16 + half * 8);
        v16bf afrag;
        #pragma unroll
        for (int i = 0; i < 8; ++i) { afrag[i] = alo[i]; afrag[i + 8] = ahi[i]; }

        #pragma unroll
        for (int j = 0; j < 4; ++j) {
            const bf16_t* bp = Bt + (size_t)(nbase + j * 16 + l16) * KDIM + k0 + half * 8;
            const v8bf blo = *(const v8bf*)(bp);
            const v8bf bhi = *(const v8bf*)(bp + 16);
            v16bf bfrag;
            #pragma unroll
            for (int i = 0; i < 8; ++i) { bfrag[i] = blo[i]; bfrag[i + 8] = bhi[i]; }
            acc[j] = __builtin_amdgcn_wmma_f32_16x16x32_bf16(
                false, afrag, false, bfrag, (short)0, acc[j], false, false);
        }
    }

    // Fused epilogue: a = acc + bias; spike = (a - VTH >= 0). TAU==1 => no v.
    #pragma unroll
    for (int j = 0; j < 4; ++j) {
        const int ocol = nbase + j * 16 + l16;
        const float bv = bsum[ocol];
        #pragma unroll
        for (int r = 0; r < 8; ++r) {
            const int orow = mbase + half * 8 + r;
            const float a  = acc[j][r] + bv;
            const float s  = (a - 1.0f >= 0.0f) ? 1.0f : 0.0f;
            if (MODE == 0) {
                ((bf16_t*)outp)[(size_t)orow * HDIM + ocol] = (bf16_t)s;
            } else {
                ((float*)outp)[((size_t)orow * TSTEPS + t) * HDIM + ocol] = s;
            }
        }
    }
}

// ---------------------------------------------------------------------------
// Delay-softmax mix + depthwise conv over T + temporal mean, fused per (n,c).
// Delay 5 shifts completely out of the T=5 window -> only w0 (d=1), w1 (d=3).
// ---------------------------------------------------------------------------
__global__ __launch_bounds__(256) void readout_kernel(
    const float* __restrict__ spk, const float* __restrict__ delay_w,
    const float* __restrict__ conv_w, float* __restrict__ ymean)
{
    const int n = blockIdx.x;
    const int c = threadIdx.x;
    const int g = c >> 5;   // 256 channels / 8 groups = 32 per group

    const float d0 = delay_w[g * 3 + 0];
    const float d1 = delay_w[g * 3 + 1];
    const float d2 = delay_w[g * 3 + 2];
    const float m  = fmaxf(d0, fmaxf(d1, d2));
    const float e0 = __expf(d0 - m), e1 = __expf(d1 - m), e2 = __expf(d2 - m);
    const float inv = 1.0f / (e0 + e1 + e2);
    const float w0 = e0 * inv;   // delay 1
    const float w1 = e1 * inv;   // delay 3

    float s[TSTEPS];
    #pragma unroll
    for (int tt = 0; tt < TSTEPS; ++tt)
        s[tt] = spk[((size_t)n * TSTEPS + tt) * HDIM + c];

    float o[TSTEPS];
    #pragma unroll
    for (int tt = 0; tt < TSTEPS; ++tt) {
        float v = s[tt];
        if (tt >= 1) v += w0 * s[tt - 1];
        if (tt >= 3) v += w1 * s[tt - 3];
        o[tt] = v;
    }

    float cw[5];
    #pragma unroll
    for (int k = 0; k < 5; ++k) cw[k] = conv_w[c * 5 + k];

    // SAME cross-correlation, K=5, then mean over T
    float y = 0.0f;
    #pragma unroll
    for (int tt = 0; tt < TSTEPS; ++tt) {
        #pragma unroll
        for (int k = 0; k < 5; ++k) {
            const int tau = tt + k - 2;
            if (tau >= 0 && tau < TSTEPS) y += cw[k] * o[tau];
        }
    }
    ymean[(size_t)n * HDIM + c] = y * 0.2f;
}

// Final small GEMM: (8192 x 256) @ (256 x 64) + bias -> d_out. ~0.27 GFLOP.
__global__ __launch_bounds__(64) void readout_gemm_kernel(
    const float* __restrict__ ymean, const float* __restrict__ W,
    const float* __restrict__ b, float* __restrict__ out)
{
    const int n = blockIdx.x;
    const int j = threadIdx.x;
    const float* yr = ymean + (size_t)n * HDIM;
    float acc = b[j];
    #pragma unroll 4
    for (int c = 0; c < HDIM; ++c)
        acc = fmaf(yr[c], W[(size_t)c * CDIM + j], acc);
    out[(size_t)n * CDIM + j] = acc;
}

// ---------------------------------------------------------------------------
extern "C" void kernel_launch(void* const* d_in, const int* in_sizes, int n_in,
                              void* d_out, int out_size, void* d_ws, size_t ws_size,
                              hipStream_t stream) {
    const float* x       = (const float*)d_in[0];
    const int*   nodes   = (const int*)  d_in[1];
    const int*   nbr1    = (const int*)  d_in[2];
    const int*   nbr2    = (const int*)  d_in[3];
    const float* Wl0     = (const float*)d_in[4];
    const float* Wr0     = (const float*)d_in[5];
    const float* bl0     = (const float*)d_in[6];
    const float* br0     = (const float*)d_in[7];
    const float* Wl1     = (const float*)d_in[8];
    const float* Wr1     = (const float*)d_in[9];
    const float* bl1     = (const float*)d_in[10];
    const float* br1     = (const float*)d_in[11];
    const float* delay_w = (const float*)d_in[12];
    const float* tsr_c   = (const float*)d_in[13];
    const float* tsr_W   = (const float*)d_in[14];
    const float* tsr_b   = (const float*)d_in[15];

    char* ws = (char*)d_ws;
    bf16_t* A0  = (bf16_t*)(ws + OFF_A0);
    bf16_t* s0  = (bf16_t*)(ws + OFF_S0);
    bf16_t* A1  = (bf16_t*)(ws + OFF_A1);
    bf16_t* Bt0 = (bf16_t*)(ws + OFF_BT0);
    bf16_t* Bt1 = (bf16_t*)(ws + OFF_BT1);
    float*  bs0 = (float*) (ws + OFF_BS0);
    float*  bs1 = (float*) (ws + OFF_BS1);
    float*  spk = (float*) (ws + OFF_SPK);
    float*  ym  = (float*) (ws + OFF_YM);

    prep_kernel<<<dim3(256), dim3(256), 0, stream>>>(
        Wl0, Wr0, Wl1, Wr1, bl0, br0, bl1, br1, Bt0, Bt1, bs0, bs1);

    for (int t = 0; t < TSTEPS; ++t) {
        gather0_kernel<<<dim3(M0ROWS), dim3(64), 0, stream>>>(
            x, nodes, nbr1, nbr2, A0, t);
        gemm_lif_kernel<0><<<dim3(M0ROWS / 64, HDIM / 64), dim3(128), 0, stream>>>(
            A0, Bt0, bs0, (void*)s0, t);
        build_a1_kernel<<<dim3(NNODES), dim3(64), 0, stream>>>(s0, A1);
        gemm_lif_kernel<1><<<dim3(NNODES / 64, HDIM / 64), dim3(128), 0, stream>>>(
            A1, Bt1, bs1, (void*)spk, t);
    }

    readout_kernel<<<dim3(NNODES), dim3(256), 0, stream>>>(spk, delay_w, tsr_c, ym);
    readout_gemm_kernel<<<dim3(NNODES), dim3(64), 0, stream>>>(ym, tsr_W, tsr_b, (float*)d_out);
}
